// Bilinear_6820408066643
// MI455X (gfx1250) — matrix operand order repaired and checked by reading, not compiled, not run on hardware
//
#include <hip/hip_runtime.h>

// Problem constants (from reference): S=8, N=2048, D=256
#define S_DIM 8
#define N_DIM 2048
#define D_DIM 256

typedef unsigned short u16;
typedef unsigned int   u32;
typedef __attribute__((ext_vector_type(16))) __bf16 v16bf;
typedef __attribute__((ext_vector_type(8)))  float  v8f;
typedef __attribute__((ext_vector_type(4)))  u32    u32x4;
typedef __attribute__((ext_vector_type(8)))  u32    u32x8;

// ---------- bf16 split helpers (RNE) ----------
__device__ __forceinline__ u16 f32_to_bf16(float f) {
  u32 u = __builtin_bit_cast(u32, f);
  u += 0x7FFFu + ((u >> 16) & 1u);   // round-to-nearest-even
  return (u16)(u >> 16);
}
__device__ __forceinline__ float bf16_to_f32(u16 h) {
  u32 u = ((u32)h) << 16;
  return __builtin_bit_cast(float, u);
}

// Split f32 array into bf16 hi (RNE) and bf16 lo (residual): x ~= hi + lo
__global__ void __launch_bounds__(256) split_bf16_kernel(
    const float* __restrict__ x, u16* __restrict__ hi, u16* __restrict__ lo, int n) {
  int i = blockIdx.x * blockDim.x + threadIdx.x;
  if (i >= n) return;
  float v = x[i];
  u16 h = f32_to_bf16(v);
  hi[i] = h;
  lo[i] = f32_to_bf16(v - bf16_to_f32(h));
}

// ---------- WMMA fragment loads (K-contiguous rows) ----------
// A 16x32 bf16 (ISA layout): lane half=0 holds K[k..k+7] and K[k+16..k+23],
// half=1 holds K[k+8..k+15] and K[k+24..k+31]  -> two 16B chunks.
__device__ __forceinline__ v16bf load_a_frag(const u16* rowK, int hl) {
  const u16* p = rowK + hl * 8;
  u32x4 c0 = *(const u32x4*)p;
  u32x4 c1 = *(const u32x4*)(p + 16);
  u32x8 c = {c0[0], c0[1], c0[2], c0[3], c1[0], c1[1], c1[2], c1[3]};
  return __builtin_bit_cast(v16bf, c);
}
// B 32x16 bf16 (same K order as A): lanes 0-15 hold K[k..k+7] and K[k+16..k+23],
// lanes 16-31 hold K[k+8..k+15] and K[k+24..k+31]  -> two 16B chunks.
__device__ __forceinline__ v16bf load_b_frag(const u16* rowK, int hl) {
  const u16* p = rowK + hl * 8;
  u32x4 c0 = *(const u32x4*)p;
  u32x4 c1 = *(const u32x4*)(p + 16);
  u32x8 c = {c0[0], c0[1], c0[2], c0[3], c1[0], c1[1], c1[2], c1[3]};
  return __builtin_bit_cast(v16bf, c);
}

// bf16x3 split product: (ah+al)*(bh+bl) ~= ah*bh + ah*bl + al*bh (f32 accum)
__device__ __forceinline__ v8f wmma3(v16bf ah, v16bf al, v16bf bh, v16bf bl, v8f acc) {
  acc = __builtin_amdgcn_wmma_f32_16x16x32_bf16(false, ah, false, bh, (short)0, acc, false, false);
  acc = __builtin_amdgcn_wmma_f32_16x16x32_bf16(false, ah, false, bl, (short)0, acc, false, false);
  acc = __builtin_amdgcn_wmma_f32_16x16x32_bf16(false, al, false, bh, (short)0, acc, false, false);
  return acc;
}

// ---------- GEMM1: Q[m,d] = sum_e T0[m,e] * K[d,e]  (M=S*N, N=D, K=D) ----------
// Block: 256 thr = 8 waves, 128x128 tile; wave: 32(M) x 64(N).
// Epilogue: split Q into bf16 hi/lo directly.
__global__ void __launch_bounds__(256) gemm1_kernel(
    const u16* __restrict__ Ah, const u16* __restrict__ Al,   // T0 split [S*N, D]
    const u16* __restrict__ Bh, const u16* __restrict__ Bl,   // Kmat split [D, D]
    u16* __restrict__ Qh, u16* __restrict__ Ql)               // out [S*N, D]
{
  const int wave = threadIdx.x >> 5;
  const int lane = threadIdx.x & 31;
  const int r    = lane & 15;
  const int hl   = lane >> 4;
  const int m0 = blockIdx.x * 128 + (wave & 3) * 32;
  const int n0 = blockIdx.y * 128 + (wave >> 2) * 64;

  const u16 *aH[2], *aL[2], *bH[4], *bL[4];
#pragma unroll
  for (int mt = 0; mt < 2; ++mt) {
    size_t row = (size_t)(m0 + mt * 16 + r) * D_DIM;
    aH[mt] = Ah + row; aL[mt] = Al + row;
  }
#pragma unroll
  for (int nt = 0; nt < 4; ++nt) {
    size_t row = (size_t)(n0 + nt * 16 + r) * D_DIM;
    bH[nt] = Bh + row; bL[nt] = Bl + row;
  }

  v8f acc[2][4];
#pragma unroll
  for (int mt = 0; mt < 2; ++mt)
#pragma unroll
    for (int nt = 0; nt < 4; ++nt) { v8f z = {}; acc[mt][nt] = z; }

  for (int k = 0; k < D_DIM; k += 32) {
    v16bf fAh[2], fAl[2], fBh[4], fBl[4];
#pragma unroll
    for (int mt = 0; mt < 2; ++mt) {
      fAh[mt] = load_a_frag(aH[mt] + k, hl);
      fAl[mt] = load_a_frag(aL[mt] + k, hl);
    }
#pragma unroll
    for (int nt = 0; nt < 4; ++nt) {
      fBh[nt] = load_b_frag(bH[nt] + k, hl);
      fBl[nt] = load_b_frag(bL[nt] + k, hl);
    }
#pragma unroll
    for (int mt = 0; mt < 2; ++mt)
#pragma unroll
      for (int nt = 0; nt < 4; ++nt)
        acc[mt][nt] = wmma3(fAh[mt], fAl[mt], fBh[nt], fBl[nt], acc[mt][nt]);
  }

  // C/D layout: lanes 0-15 -> N=lane, M=p; lanes 16-31 -> N=lane-16, M=p+8
#pragma unroll
  for (int mt = 0; mt < 2; ++mt)
#pragma unroll
    for (int nt = 0; nt < 4; ++nt)
#pragma unroll
      for (int p = 0; p < 8; ++p) {
        int m = m0 + mt * 16 + hl * 8 + p;
        int n = n0 + nt * 16 + r;
        float x = acc[mt][nt][p];
        u16 h = f32_to_bf16(x);
        size_t idx = (size_t)m * D_DIM + n;
        Qh[idx] = h;
        Ql[idx] = f32_to_bf16(x - bf16_to_f32(h));
      }
}

// ---------- GEMM2: out[s,i,j] = sum_d T1[s,i,d] * Q[s,j,d] + bias ----------
// Per batch s: [N,D] x [D,N] -> [N,N]. Same 128x128 block / 32x64 wave tiling.
__global__ void __launch_bounds__(256) gemm2_kernel(
    const u16* __restrict__ Ah, const u16* __restrict__ Al,   // T1 split [S*N, D]
    const u16* __restrict__ Bh, const u16* __restrict__ Bl,   // Q  split [S*N, D]
    const float* __restrict__ bias, float* __restrict__ out)
{
  const int s    = blockIdx.z;
  const int wave = threadIdx.x >> 5;
  const int lane = threadIdx.x & 31;
  const int r    = lane & 15;
  const int hl   = lane >> 4;
  const int m0 = blockIdx.x * 128 + (wave & 3) * 32;
  const int n0 = blockIdx.y * 128 + (wave >> 2) * 64;
  const size_t sBase = (size_t)s * N_DIM;

  const u16 *aH[2], *aL[2], *bH[4], *bL[4];
#pragma unroll
  for (int mt = 0; mt < 2; ++mt) {
    size_t row = (sBase + (size_t)(m0 + mt * 16 + r)) * D_DIM;
    aH[mt] = Ah + row; aL[mt] = Al + row;
  }
#pragma unroll
  for (int nt = 0; nt < 4; ++nt) {
    size_t row = (sBase + (size_t)(n0 + nt * 16 + r)) * D_DIM;
    bH[nt] = Bh + row; bL[nt] = Bl + row;
  }

  v8f acc[2][4];
#pragma unroll
  for (int mt = 0; mt < 2; ++mt)
#pragma unroll
    for (int nt = 0; nt < 4; ++nt) { v8f z = {}; acc[mt][nt] = z; }

  for (int k = 0; k < D_DIM; k += 32) {
    v16bf fAh[2], fAl[2], fBh[4], fBl[4];
#pragma unroll
    for (int mt = 0; mt < 2; ++mt) {
      fAh[mt] = load_a_frag(aH[mt] + k, hl);
      fAl[mt] = load_a_frag(aL[mt] + k, hl);
    }
#pragma unroll
    for (int nt = 0; nt < 4; ++nt) {
      fBh[nt] = load_b_frag(bH[nt] + k, hl);
      fBl[nt] = load_b_frag(bL[nt] + k, hl);
    }
#pragma unroll
    for (int mt = 0; mt < 2; ++mt)
#pragma unroll
      for (int nt = 0; nt < 4; ++nt)
        acc[mt][nt] = wmma3(fAh[mt], fAl[mt], fBh[nt], fBl[nt], acc[mt][nt]);
  }

  const float bv = bias[0];
#pragma unroll
  for (int mt = 0; mt < 2; ++mt)
#pragma unroll
    for (int nt = 0; nt < 4; ++nt)
#pragma unroll
      for (int p = 0; p < 8; ++p) {
        int m = m0 + mt * 16 + hl * 8 + p;
        int n = n0 + nt * 16 + r;
        out[(size_t)s * N_DIM * N_DIM + (size_t)m * N_DIM + n] = acc[mt][nt][p] + bv;
      }
}

extern "C" void kernel_launch(void* const* d_in, const int* in_sizes, int n_in,
                              void* d_out, int out_size, void* d_ws, size_t ws_size,
                              hipStream_t stream) {
  (void)in_sizes; (void)n_in; (void)out_size; (void)ws_size;
  const float* t0   = (const float*)d_in[0];   // [S,N,D]
  const float* t1   = (const float*)d_in[1];   // [S,N,D]
  const float* km   = (const float*)d_in[2];   // [D,D]
  const float* bias = (const float*)d_in[3];   // [1]
  float* out = (float*)d_out;                  // [S,N,N]

  const size_t EL = (size_t)S_DIM * N_DIM * D_DIM;     // 4,194,304
  const size_t KE = (size_t)D_DIM * D_DIM;             // 65,536
  u16* T0h = (u16*)d_ws;
  u16* T0l = T0h + EL;
  u16* T1h = T0l + EL;
  u16* T1l = T1h + EL;
  u16* Qh  = T1l + EL;
  u16* Ql  = Qh  + EL;
  u16* Kh  = Ql  + EL;
  u16* Kl  = Kh  + KE;
  // total workspace: (6*EL + 2*KE)*2 bytes ~= 50.6 MB

  split_bf16_kernel<<<(unsigned)(EL / 256), 256, 0, stream>>>(t0, T0h, T0l, (int)EL);
  split_bf16_kernel<<<(unsigned)(EL / 256), 256, 0, stream>>>(t1, T1h, T1l, (int)EL);
  split_bf16_kernel<<<(unsigned)(KE / 256), 256, 0, stream>>>(km, Kh, Kl, (int)KE);

  dim3 g1(S_DIM * N_DIM / 128, D_DIM / 128, 1);        // (128, 2)
  gemm1_kernel<<<g1, 256, 0, stream>>>(T0h, T0l, Kh, Kl, Qh, Ql);

  dim3 g2(N_DIM / 128, N_DIM / 128, S_DIM);            // (16, 16, 8)
  gemm2_kernel<<<g2, 256, 0, stream>>>(T1h, T1l, Qh, Ql, bias, out);
}
